// SceneGraphGATLayer_11209864642747
// MI455X (gfx1250) — matrix operand order repaired and checked by reading, not compiled, not run on hardware
//
#include <hip/hip_runtime.h>
#include <cstdint>
#include <cstddef>

// ---------------------------------------------------------------------------
// SceneGraphGATLayer for MI455X (gfx1250): bf16 WMMA GEMM pipeline.
// B=32, N=256, D=1024, H=16, E=512, ED=256, HD=64, FF=4096.
// All matmuls run on v_wmma_f32_16x16x32_bf16; B operands are pre-transposed
// so every fragment load is a contiguous b128 (no scalar LDS/global gathers).
// ---------------------------------------------------------------------------

typedef __bf16 bf16_t;
typedef __attribute__((ext_vector_type(16))) __bf16 v16bf;
typedef __attribute__((ext_vector_type(8)))  float  v8f;

constexpr int cB = 32, cN = 256, cD = 1024, cH = 16, cE = 512, cED = 256, cHD = 64, cFF = 4096;
constexpr int cM = cB * cN; // 8192 rows
constexpr float kNEG = -3.40282346638528859811704183485e38f;

// ---------------------------------------------------------------------------
// f32 -> bf16 convert + transpose (weights): out[n*K + k] = (bf16)in[k*N + n]
// ---------------------------------------------------------------------------
__global__ __launch_bounds__(256)
void cvt_transpose_kernel(const float* __restrict__ in, bf16_t* __restrict__ out,
                          int K, int N) {
    __shared__ bf16_t t[32][33];
    const int n0 = blockIdx.x * 32, k0 = blockIdx.y * 32;
    const int j = threadIdx.x & 31, i0 = threadIdx.x >> 5;   // i0 = 0..7
    #pragma unroll
    for (int p = 0; p < 4; ++p) {
        const int i = i0 + p * 8;
        t[j][i] = (bf16_t)in[(size_t)(k0 + i) * N + n0 + j];  // t[n_loc][k_loc]
    }
    __syncthreads();
    #pragma unroll
    for (int p = 0; p < 4; ++p) {
        const int i = i0 + p * 8;
        out[(size_t)(n0 + i) * K + k0 + j] = t[i][j];
    }
}

// ---------------------------------------------------------------------------
// LayerNorm over D=1024, one block (256 threads) per row, bf16 output
// ---------------------------------------------------------------------------
__global__ __launch_bounds__(256)
void ln_bf16_kernel(const float* __restrict__ x, const float* __restrict__ g,
                    const float* __restrict__ bb, bf16_t* __restrict__ out) {
    const int row = blockIdx.x;
    const float* xr = x + (size_t)row * cD;
    const int tid = threadIdx.x;
    float s = 0.f, s2 = 0.f;
    for (int i = tid; i < cD; i += 256) { float v = xr[i]; s += v; s2 += v * v; }
    __shared__ float r1[256], r2[256];
    r1[tid] = s; r2[tid] = s2;
    __syncthreads();
    for (int off = 128; off > 0; off >>= 1) {
        if (tid < off) { r1[tid] += r1[tid + off]; r2[tid] += r2[tid + off]; }
        __syncthreads();
    }
    const float mean = r1[0] * (1.f / cD);
    const float var  = r2[0] * (1.f / cD) - mean * mean;
    const float rstd = rsqrtf(var + 1e-5f);
    bf16_t* orow = out + (size_t)row * cD;
    for (int i = tid; i < cD; i += 256)
        orow[i] = (bf16_t)((xr[i] - mean) * rstd * g[i] + bb[i]);
}

// ---------------------------------------------------------------------------
// bf16 WMMA GEMM: C[M,N] = A[M,K] * Bt^T  (Bt is [N,K], pre-transposed)
//   mode 0: outB = bf16(acc + bias)                      (Q/K projections)
//   mode 1: outF = acc + bias + res                      (Wo + residual)
//   mode 2: outB = bf16(silu(acc + bias))                (W1 + SiLU)
//   mode 3: outF = (acc + bias + res) * rowmask[row]     (W2 + residual + mask)
//   mode 4: outB transposed per batch: vt[(b*D+col)*N + n] (V projection)
// Block tile 128x128, 8 waves (4x2), each wave -> 32x64 (2x4 WMMA tiles):
// B-fragments reused across both A-fragments -> 12 ds_load_b128 per 8 WMMAs.
// Software-pipelined global->LDS staging.
// ---------------------------------------------------------------------------
#define GBM 128
#define GBN 128
#define GBK 32
#define GBK_PAD 48   // 96B row stride: b128-aligned, breaks bank conflicts

__global__ __launch_bounds__(256)
void gemm_bf16_kernel(const bf16_t* __restrict__ A, const bf16_t* __restrict__ Bt,
                      const float* __restrict__ bias, const float* __restrict__ res,
                      const float* __restrict__ rmask,
                      float* __restrict__ outF, bf16_t* __restrict__ outB,
                      int M, int N, int K, int mode) {
    __shared__ bf16_t As[GBM][GBK];        // 8 KB   (rows = m)
    __shared__ bf16_t Bs[GBN][GBK_PAD];    // 12 KB  (rows = n, pre-transposed)

    const int tid  = threadIdx.x;
    const int wave = tid >> 5, lane = tid & 31;
    const int bm0 = blockIdx.y * GBM, bn0 = blockIdx.x * GBN;
    const int wy = wave >> 1;            // 0..3 : 32-row band
    const int wx = wave & 1;             // 0..1 : 64-col half
    const int lm   = lane & 15;          // A row / B col within fragment
    const int kh8  = (lane >> 4) * 8;    // A-frag K base (two runs of 8)
    const int kh16 = (lane >> 4) * 16;   // unused: B-frag also uses two kh8 runs

    // staging coords: 512 chunks of 8 elems for each of A and B, 2/thread
    const int sr = tid >> 1, sc = (tid & 1) * 16;

    v8f acc[2][4] = {};
    uint4 ra0, ra1, rb0, rb1;

    // prologue: stage k0 = 0
    ra0 = *reinterpret_cast<const uint4*>(A  + (size_t)(bm0 + sr) * K + sc);
    ra1 = *reinterpret_cast<const uint4*>(A  + (size_t)(bm0 + sr) * K + sc + 8);
    rb0 = *reinterpret_cast<const uint4*>(Bt + (size_t)(bn0 + sr) * K + sc);
    rb1 = *reinterpret_cast<const uint4*>(Bt + (size_t)(bn0 + sr) * K + sc + 8);

    for (int k0 = 0; k0 < K; k0 += GBK) {
        *reinterpret_cast<uint4*>(&As[sr][sc])     = ra0;
        *reinterpret_cast<uint4*>(&As[sr][sc + 8]) = ra1;
        *reinterpret_cast<uint4*>(&Bs[sr][sc])     = rb0;
        *reinterpret_cast<uint4*>(&Bs[sr][sc + 8]) = rb1;
        __syncthreads();

        if (k0 + GBK < K) {  // prefetch next tile into registers during compute
            const int kn = k0 + GBK;
            ra0 = *reinterpret_cast<const uint4*>(A  + (size_t)(bm0 + sr) * K + kn + sc);
            ra1 = *reinterpret_cast<const uint4*>(A  + (size_t)(bm0 + sr) * K + kn + sc + 8);
            rb0 = *reinterpret_cast<const uint4*>(Bt + (size_t)(bn0 + sr) * K + kn + sc);
            rb1 = *reinterpret_cast<const uint4*>(Bt + (size_t)(bn0 + sr) * K + kn + sc + 8);
        }

        // two A fragments (M sub-tiles), each two contiguous 8-elem K runs
        v16bf afr[2];
        #pragma unroll
        for (int ms = 0; ms < 2; ++ms) {
            const bf16_t* ap = &As[wy * 32 + ms * 16 + lm][0];
            *reinterpret_cast<uint4*>(&afr[ms]) =
                *reinterpret_cast<const uint4*>(ap + kh8);
            *(reinterpret_cast<uint4*>(&afr[ms]) + 1) =
                *reinterpret_cast<const uint4*>(ap + 16 + kh8);
        }

        #pragma unroll
        for (int t = 0; t < 4; ++t) {
            // B fragment: two 8-elem K runs (kh8, 16+kh8) at column (lane&15)
            const bf16_t* bp = &Bs[wx * 64 + t * 16 + lm][kh8];
            v16bf bfr;
            *reinterpret_cast<uint4*>(&bfr)       = *reinterpret_cast<const uint4*>(bp);
            *(reinterpret_cast<uint4*>(&bfr) + 1) = *reinterpret_cast<const uint4*>(bp + 16);
            #pragma unroll
            for (int ms = 0; ms < 2; ++ms)  // reuse bfr across both A fragments
                acc[ms][t] = __builtin_amdgcn_wmma_f32_16x16x32_bf16(
                    false, afr[ms], false, bfr, (short)0, acc[ms][t], false, false);
        }
        __syncthreads();
    }

    // ---- epilogue ----
    #pragma unroll
    for (int ms = 0; ms < 2; ++ms) {
        const int mbase = bm0 + wy * 32 + ms * 16 + (lane >> 4) * 8;
        if (mode == 4) {   // transposed bf16 output: vt[(batch*D + col)*N + n]
            const int batch = mbase >> 8, nbase = mbase & (cN - 1);
            #pragma unroll
            for (int t = 0; t < 4; ++t) {
                const int col = bn0 + wx * 64 + t * 16 + lm;
                const float bcol = bias[col];
                union { uint4 u; bf16_t h[8]; } pk;
                #pragma unroll
                for (int r = 0; r < 8; ++r) pk.h[r] = (bf16_t)(acc[ms][t][r] + bcol);
                *reinterpret_cast<uint4*>(outB + ((size_t)batch * cD + col) * cN + nbase) = pk.u;
            }
            continue;
        }
        #pragma unroll
        for (int t = 0; t < 4; ++t) {
            const int col = bn0 + wx * 64 + t * 16 + lm;
            const float bcol = bias[col];
            #pragma unroll
            for (int r = 0; r < 8; ++r) {
                const int row = mbase + r;
                const size_t idx = (size_t)row * N + col;
                float v = acc[ms][t][r] + bcol;
                if (mode == 0) {
                    outB[idx] = (bf16_t)v;
                } else if (mode == 1) {
                    outF[idx] = v + res[idx];
                } else if (mode == 2) {
                    outB[idx] = (bf16_t)(v / (1.f + __expf(-v)));   // SiLU
                } else {
                    outF[idx] = (v + res[idx]) * rmask[row];
                }
            }
        }
    }
}

// ---------------------------------------------------------------------------
// Edge map init: -1 = no relation, -2 = diagonal (self edge, zero bias)
// ---------------------------------------------------------------------------
__global__ __launch_bounds__(256)
void edge_map_init_kernel(int* __restrict__ emap) {
    const int idx = blockIdx.x * 256 + threadIdx.x;   // over B*N*N = 2M
    const int i = (idx >> 8) & (cN - 1);
    const int j = idx & (cN - 1);
    emap[idx] = (i == j) ? -2 : -1;
}

// ---------------------------------------------------------------------------
// Per-edge bias head projection + symmetric scatter of edge ids
// ---------------------------------------------------------------------------
__global__ __launch_bounds__(256)
void edge_scatter_kernel(const long long* __restrict__ edges,
                         const float* __restrict__ emask,
                         const float* __restrict__ emb,
                         const float* __restrict__ We, const float* __restrict__ be,
                         float* __restrict__ eb, int* __restrict__ emap) {
    const int beidx = blockIdx.x;            // b*E + e
    const int b = beidx / cE, e = beidx % cE;
    __shared__ float semb[cED];
    if (threadIdx.x < cED)
        semb[threadIdx.x] = emb[(size_t)beidx * cED + threadIdx.x];
    __syncthreads();
    const int h = threadIdx.x;
    if (h < cH) {
        float acc = be[h];
        for (int j = 0; j < cED; ++j) acc += semb[j] * We[j * cH + h];
        eb[(size_t)beidx * cH + h] = acc;
    }
    if (threadIdx.x == 0 && emask[beidx] != 0.f) {
        long long s = edges[(size_t)beidx * 2 + 0];
        long long d = edges[(size_t)beidx * 2 + 1];
        int si = (int)(s < 0 ? 0 : (s > cN - 1 ? cN - 1 : s));
        int di = (int)(d < 0 ? 0 : (d > cN - 1 ? cN - 1 : d));
        emap[((size_t)b * cN + si) * cN + di] = e;
        emap[((size_t)b * cN + di) * cN + si] = e;
    }
}

// ---------------------------------------------------------------------------
// Attention: one workgroup per (b, h, 16-query tile).
// S = Q K^T via WMMA (K rows are contiguous b128 runs = transposed-B layout),
// bias/mask/softmax in LDS, O = P V via WMMA with V pre-transposed (vt).
// ---------------------------------------------------------------------------
__global__ __launch_bounds__(256)
void attn_kernel(const bf16_t* __restrict__ q, const bf16_t* __restrict__ k,
                 const bf16_t* __restrict__ vt,
                 const float* __restrict__ eb, const int* __restrict__ emap,
                 const float* __restrict__ nmask, bf16_t* __restrict__ out) {
    const int qt = blockIdx.x;               // query tile 0..15
    const int h  = blockIdx.y;
    const int b  = blockIdx.z;
    const int q0 = qt * 16;
    const int hc = h * cHD;

    __shared__ float  S[16][272];            // padded scores
    __shared__ bf16_t P[16][256];            // softmax probs (bf16)
    __shared__ float  red[16][17];
    __shared__ float  rowstat[16];

    const int tid  = threadIdx.x;
    const int wave = tid >> 5, lane = tid & 31;
    const int lm   = lane & 15;
    const int kh8  = (lane >> 4) * 8;
    const int kh16 = (lane >> 4) * 16;
    const size_t rowQ = (size_t)b * cN + q0;

    // ---- S = q . k^T  (each wave: 2 key tiles of 16; Q frag hoisted) ----
    {
        v8f acc2[2] = {};
        #pragma unroll
        for (int k0 = 0; k0 < cHD; k0 += 32) {
            v16bf afr;
            const bf16_t* qp = q + (rowQ + lm) * cD + hc + k0;
            *reinterpret_cast<uint4*>(&afr) =
                *reinterpret_cast<const uint4*>(qp + kh8);
            *(reinterpret_cast<uint4*>(&afr) + 1) =
                *reinterpret_cast<const uint4*>(qp + 16 + kh8);
            #pragma unroll
            for (int tt = 0; tt < 2; ++tt) {
                const int key0 = (wave * 2 + tt) * 16;
                // B[kk][n] = k[key0+n][hc+k0+kk]: two 8-elem runs per lane
                const bf16_t* kp = k + ((size_t)b * cN + key0 + lm) * cD + hc + k0 + kh8;
                v16bf bfr;
                *reinterpret_cast<uint4*>(&bfr)       = *reinterpret_cast<const uint4*>(kp);
                *(reinterpret_cast<uint4*>(&bfr) + 1) = *reinterpret_cast<const uint4*>(kp + 16);
                acc2[tt] = __builtin_amdgcn_wmma_f32_16x16x32_bf16(
                    false, afr, false, bfr, (short)0, acc2[tt], false, false);
            }
        }
        const int mb = (lane >> 4) * 8;
        #pragma unroll
        for (int tt = 0; tt < 2; ++tt) {
            const int col = (wave * 2 + tt) * 16 + lm;
            #pragma unroll
            for (int r = 0; r < 8; ++r) S[mb + r][col] = acc2[tt][r] * 0.125f; // 1/sqrt(64)
        }
    }
    __syncthreads();

    // ---- relation bias + adjacency/node masking ----
    for (int idx = tid; idx < 16 * 256; idx += 256) {
        const int i = idx >> 8, j = idx & 255;
        const int id = emap[((size_t)b * cN + q0 + i) * cN + j];
        const float bias = (id >= 0) ? eb[((size_t)b * cE + id) * cH + h] : 0.f;
        const bool ok = (id != -1) && (nmask[b * cN + q0 + i] != 0.f)
                                   && (nmask[b * cN + j] != 0.f);
        S[i][j] = ok ? (S[i][j] + bias) : kNEG;
    }
    __syncthreads();

    // ---- softmax: 16 threads per row ----
    {
        const int r = tid >> 4, c = tid & 15;
        float mx = kNEG;
        for (int j = c; j < 256; j += 16) mx = fmaxf(mx, S[r][j]);
        red[r][c] = mx;
        __syncthreads();
        if (c == 0) { float m = red[r][0]; for (int j = 1; j < 16; ++j) m = fmaxf(m, red[r][j]); rowstat[r] = m; }
        __syncthreads();
        const float rm = rowstat[r];
        float s = 0.f;
        for (int j = c; j < 256; j += 16) { float e_ = __expf(S[r][j] - rm); S[r][j] = e_; s += e_; }
        red[r][c] = s;
        __syncthreads();
        if (c == 0) { float t = 0.f; for (int j = 0; j < 16; ++j) t += red[r][j]; rowstat[r] = t; }
        __syncthreads();
        const float inv = 1.f / rowstat[r];
        for (int j = c; j < 256; j += 16) P[r][j] = (bf16_t)(S[r][j] * inv);
    }
    __syncthreads();

    // ---- O = P . V^T-layout  (waves 0..3: one 16x16 output tile each) ----
    if (wave < 4) {
        v8f acc = {};
        const int nc = hc + wave * 16;
        #pragma unroll
        for (int k0 = 0; k0 < cN; k0 += 32) {
            v16bf afr, bfr;
            const bf16_t* pp = &P[lm][k0];
            *reinterpret_cast<uint4*>(&afr) =
                *reinterpret_cast<const uint4*>(pp + kh8);
            *(reinterpret_cast<uint4*>(&afr) + 1) =
                *reinterpret_cast<const uint4*>(pp + 16 + kh8);
            // B[kk][n] = V[k0+kk][nc+n] = vt[(b*D + nc+n)*N + k0+kk]: contiguous
            const bf16_t* vp = vt + ((size_t)b * cD + nc + lm) * cN + k0 + kh8;
            *reinterpret_cast<uint4*>(&bfr)       = *reinterpret_cast<const uint4*>(vp);
            *(reinterpret_cast<uint4*>(&bfr) + 1) = *reinterpret_cast<const uint4*>(vp + 16);
            acc = __builtin_amdgcn_wmma_f32_16x16x32_bf16(
                false, afr, false, bfr, (short)0, acc, false, false);
        }
        const int col = nc + lm, mb = (lane >> 4) * 8;
        #pragma unroll
        for (int r = 0; r < 8; ++r)
            out[(rowQ + mb + r) * cD + col] = (bf16_t)acc[r];
    }
}

// ---------------------------------------------------------------------------
// Host-side orchestration
// ---------------------------------------------------------------------------
extern "C" void kernel_launch(void* const* d_in, const int* in_sizes, int n_in,
                              void* d_out, int out_size, void* d_ws, size_t ws_size,
                              hipStream_t stream) {
    (void)in_sizes; (void)n_in; (void)out_size; (void)ws_size;

    const float*     x      = (const float*)d_in[0];
    const float*     nmask  = (const float*)d_in[1];
    const long long* edges  = (const long long*)d_in[2];
    const float*     emask  = (const float*)d_in[3];
    const float*     emb    = (const float*)d_in[4];
    const float*     ln_g   = (const float*)d_in[5];
    const float*     ln_b   = (const float*)d_in[6];
    const float*     Wq = (const float*)d_in[7];  const float* bq = (const float*)d_in[8];
    const float*     Wk = (const float*)d_in[9];  const float* bk = (const float*)d_in[10];
    const float*     Wv = (const float*)d_in[11]; const float* bv = (const float*)d_in[12];
    const float*     We = (const float*)d_in[13]; const float* be = (const float*)d_in[14];
    const float*     Wo = (const float*)d_in[15]; const float* bo = (const float*)d_in[16];
    const float*     ff_g = (const float*)d_in[17]; const float* ff_b = (const float*)d_in[18];
    const float*     W1 = (const float*)d_in[19]; const float* b1 = (const float*)d_in[20];
    const float*     W2 = (const float*)d_in[21]; const float* b2 = (const float*)d_in[22];

    char* ws = (char*)d_ws;
    size_t off = 0;
    auto alloc = [&](size_t bytes) -> void* {
        void* p = ws + off;
        off = (off + bytes + 255) & ~(size_t)255;
        return p;
    };

    bf16_t* wq_t = (bf16_t*)alloc((size_t)cD * cD * 2);
    bf16_t* wk_t = (bf16_t*)alloc((size_t)cD * cD * 2);
    bf16_t* wv_t = (bf16_t*)alloc((size_t)cD * cD * 2);
    bf16_t* wo_t = (bf16_t*)alloc((size_t)cD * cD * 2);
    bf16_t* w1_t = (bf16_t*)alloc((size_t)cD * cFF * 2);
    bf16_t* w2_t = (bf16_t*)alloc((size_t)cFF * cD * 2);
    bf16_t* xn_b = (bf16_t*)alloc((size_t)cM * cD * 2);
    bf16_t* q_b  = (bf16_t*)alloc((size_t)cM * cD * 2);
    bf16_t* k_b  = (bf16_t*)alloc((size_t)cM * cD * 2);
    bf16_t* vt_b = (bf16_t*)alloc((size_t)cB * cD * cN * 2);  // V transposed/batch
    bf16_t* ao_b = (bf16_t*)alloc((size_t)cM * cD * 2);
    bf16_t* h_b  = (bf16_t*)alloc((size_t)cM * cD * 2);
    bf16_t* h1_b = (bf16_t*)alloc((size_t)cM * cFF * 2);
    float*  x2_f = (float*)alloc((size_t)cM * cD * 4);
    int*    emap = (int*)alloc((size_t)cB * cN * cN * 4);
    float*  ebuf = (float*)alloc((size_t)cB * cE * cH * 4);

    // 1. weights -> bf16, transposed to [N,K] for the GEMM B path
    {
        dim3 gDD(cD / 32, cD / 32);
        cvt_transpose_kernel<<<gDD, 256, 0, stream>>>(Wq, wq_t, cD, cD);
        cvt_transpose_kernel<<<gDD, 256, 0, stream>>>(Wk, wk_t, cD, cD);
        cvt_transpose_kernel<<<gDD, 256, 0, stream>>>(Wv, wv_t, cD, cD);
        cvt_transpose_kernel<<<gDD, 256, 0, stream>>>(Wo, wo_t, cD, cD);
        dim3 g1(cFF / 32, cD / 32);   // W1: [D,FF] -> [FF,D]
        cvt_transpose_kernel<<<g1, 256, 0, stream>>>(W1, w1_t, cD, cFF);
        dim3 g2(cD / 32, cFF / 32);   // W2: [FF,D] -> [D,FF]
        cvt_transpose_kernel<<<g2, 256, 0, stream>>>(W2, w2_t, cFF, cD);
    }

    // 2. LN1 -> xn (bf16)
    ln_bf16_kernel<<<cM, 256, 0, stream>>>(x, ln_g, ln_b, xn_b);

    // 3. Q/K/V projections
    {
        dim3 grid(cD / GBN, cM / GBM);
        gemm_bf16_kernel<<<grid, 256, 0, stream>>>(xn_b, wq_t, bq, nullptr, nullptr,
                                                   nullptr, q_b, cM, cD, cD, 0);
        gemm_bf16_kernel<<<grid, 256, 0, stream>>>(xn_b, wk_t, bk, nullptr, nullptr,
                                                   nullptr, k_b, cM, cD, cD, 0);
        gemm_bf16_kernel<<<grid, 256, 0, stream>>>(xn_b, wv_t, bv, nullptr, nullptr,
                                                   nullptr, vt_b, cM, cD, cD, 4);
    }

    // 4. edge map + per-edge head bias
    edge_map_init_kernel<<<(cB * cN * cN) / 256, 256, 0, stream>>>(emap);
    edge_scatter_kernel<<<cB * cE, 256, 0, stream>>>(edges, emask, emb, We, be, ebuf, emap);

    // 5. attention
    {
        dim3 grid(cN / 16, cH, cB);
        attn_kernel<<<grid, 256, 0, stream>>>(q_b, k_b, vt_b, ebuf, emap, nmask, ao_b);
    }

    // 6. Wo projection + residual -> x2 (f32)
    {
        dim3 grid(cD / GBN, cM / GBM);
        gemm_bf16_kernel<<<grid, 256, 0, stream>>>(ao_b, wo_t, bo, x, nullptr,
                                                   x2_f, nullptr, cM, cD, cD, 1);
    }

    // 7. LN2 -> h (bf16)
    ln_bf16_kernel<<<cM, 256, 0, stream>>>(x2_f, ff_g, ff_b, h_b);

    // 8. FFN up + SiLU -> h1 (bf16)
    {
        dim3 grid(cFF / GBN, cM / GBM);
        gemm_bf16_kernel<<<grid, 256, 0, stream>>>(h_b, w1_t, b1, nullptr, nullptr,
                                                   nullptr, h1_b, cM, cFF, cD, 2);
    }

    // 9. FFN down + residual + node mask -> d_out (f32)
    {
        dim3 grid(cD / GBN, cM / GBM);
        gemm_bf16_kernel<<<grid, 256, 0, stream>>>(h1_b, w2_t, b2, x2_f, nmask,
                                                   (float*)d_out, nullptr, cM, cD, cFF, 3);
    }
}
